// FAM_8950711846062
// MI455X (gfx1250) — matrix operand order repaired and checked by reading, not compiled, not run on hardware
//
#include <hip/hip_runtime.h>

typedef _Float16 v8h  __attribute__((ext_vector_type(8)));
typedef _Float16 v16h __attribute__((ext_vector_type(16)));
typedef float    v8f  __attribute__((ext_vector_type(8)));
typedef float    v4f  __attribute__((ext_vector_type(4)));

#define BN   8
#define HH   64
#define WW   64
#define CX   128
#define CU   128
#define CO   128
#define KOFF 18
#define K1   2304          // offset conv reduction (256*9)
#define K2   1152          // main conv reduction (128*9)
#define KB1  (K1/32)       // 72
#define KB2  (K2/32)       // 36
#define T1   2             // N tiles, offset conv (18 -> 32)
#define T2   8             // N tiles, main conv (128)

#define OFFS_FLOATS (BN*KOFF*HH*WW)   // 589824 f32
#define FRAG1_HALFS (T1*KB1*32*16)    // 73728 f16
#define FRAG2_HALFS (T2*KB2*32*16)    // 147456 f16

#define PLANE (HH*WW)

__device__ __forceinline__ int iclamp(int v, int lo, int hi) {
    return v < lo ? lo : (v > hi ? hi : v);
}

__device__ __forceinline__ v16h make_a(const _Float16* p) {
    v8h alo = *(const v8h*)(p);
    v8h ahi = *(const v8h*)(p + 16);
    return __builtin_shufflevector(alo, ahi,
        0, 1, 2, 3, 4, 5, 6, 7, 8, 9, 10, 11, 12, 13, 14, 15);
}

__device__ __forceinline__ void store_tile(float* op, const v8f& acc, float bv) {
    v4f lo = { acc[0] + bv, acc[1] + bv, acc[2] + bv, acc[3] + bv };
    v4f hi = { acc[4] + bv, acc[5] + bv, acc[6] + bv, acc[7] + bv };
    *(v4f*)(op)     = lo;
    *(v4f*)(op + 4) = hi;
}

// ---------------------------------------------------------------------------
// Pre-swizzle both weight matrices into WMMA B-fragment layout (f16).
// Fragment per lane: element i -> K = kb*32 + (lane<16?0:8) + (i<8 ? i : i+8),
// column n = t*16 + (lane&15). Zero-pad n >= Nreal.
// ---------------------------------------------------------------------------
__global__ __launch_bounds__(256) void fam_prep_weights(
    const float* __restrict__ weight,    // (128,128,3,3) -> B[k][n]=weight[n*1152+k]
    const float* __restrict__ offset_w,  // (18,256,3,3)  -> B[k][n]=offset_w[n*2304+k]
    _Float16* __restrict__ frag1,
    _Float16* __restrict__ frag2) {
    int idx = blockIdx.x * 256 + threadIdx.x;
    const float* src;
    _Float16* dst;
    int KB, Nreal, Ksz, li;
    if (idx < FRAG1_HALFS) {
        src = offset_w; dst = frag1; KB = KB1; Nreal = KOFF; Ksz = K1; li = idx;
    } else {
        src = weight;   dst = frag2; KB = KB2; Nreal = CO;   Ksz = K2; li = idx - FRAG1_HALFS;
    }
    int i    = li & 15;
    int lane = (li >> 4) & 31;
    int kb   = (li >> 9) % KB;
    int t    = (li >> 9) / KB;
    int k = kb * 32 + ((lane & 16) ? 8 : 0) + (i < 8 ? i : i + 8);
    int n = t * 16 + (lane & 15);
    float v = (n < Nreal) ? src[n * Ksz + k] : 0.0f;
    dst[li] = (_Float16)v;
}

// ---------------------------------------------------------------------------
// Offset conv: 3x3, Cin=256 (concat x,u), Cout=18, zero pad=1.
// M=32 pixels per workgroup (2 waves; each wave: 1 N-tile x 2 M-tiles so the
// B fragment is register-reused). K processed in four 576-wide chunks (36 KB LDS).
// ---------------------------------------------------------------------------
__global__ __launch_bounds__(64) void fam_offset_conv(
    const float* __restrict__ x,
    const float* __restrict__ u,
    const float* __restrict__ offset_b,
    const _Float16* __restrict__ frag1,
    float* __restrict__ offs) {
    __shared__ __align__(16) _Float16 Ash[32 * 576];   // 36 KB

    int tile = blockIdx.x;                  // 1024 tiles
    int x0 = (tile & 1) * 32;
    int y  = (tile >> 1) & (HH - 1);
    int b  = tile >> 7;
    int tid   = threadIdx.x;
    int lane  = tid & 31;
    int wv    = tid >> 5;                   // N-tile (0,1)
    int m     = lane & 15;
    int aoff  = (lane & 16) ? 8 : 0;

    int p     = tid >> 1;                   // pixel 0..31 (A-build role)
    int lane2 = tid & 1;
    int xb    = x0 + p;

    v8f acc0 = {}, acc1 = {};               // M-tile 0 (px0-15), M-tile 1 (px16-31)
    const _Float16* bb = frag1 + (size_t)wv * KB1 * 512;

    for (int icq = 0; icq < 4; ++icq) {     // 64 input channels per chunk
        __syncthreads();
        // division-free im2col: thread owns pixel p, channels lane2+2*cc
        for (int cc = 0; cc < 32; ++cc) {
            int icl = lane2 + cc * 2;       // 0..63 within chunk
            int ic  = icq * 64 + icl;       // 0..255 global
            const float* src = (ic < CX)
                ? (x + (size_t)(b * CX + ic) * PLANE)
                : (u + (size_t)(b * CU + (ic - CX)) * PLANE);
            _Float16* arow = Ash + p * 576 + icl * 9;
#pragma unroll
            for (int ky = 0; ky < 3; ++ky) {
                int yy = y + ky - 1;
                bool yv = (yy >= 0) && (yy < HH);
                const float* rp = src + yy * WW;
#pragma unroll
                for (int kx = 0; kx < 3; ++kx) {
                    int xx = xb + kx - 1;
                    float v = (yv && xx >= 0 && xx < WW) ? rp[xx] : 0.0f;
                    arow[ky * 3 + kx] = (_Float16)v;
                }
            }
        }
        __syncthreads();

        for (int kb2 = 0; kb2 < 18; ++kb2) {
            int kbg = icq * 18 + kb2;
            int ab  = m * 576 + kb2 * 32 + aoff;
            v16h a0 = make_a(Ash + ab);
            v16h a1 = make_a(Ash + ab + 16 * 576);
            v16h bf = *(const v16h*)(bb + ((size_t)kbg * 32 + lane) * 16);
            acc0 = __builtin_amdgcn_wmma_f32_16x16x32_f16(
                false, a0, false, bf, (short)0, acc0, false, false);
            acc1 = __builtin_amdgcn_wmma_f32_16x16x32_f16(
                false, a1, false, bf, (short)0, acc1, false, false);
        }
    }

    int n  = lane & 15;
    int ch = wv * 16 + n;
    if (ch < KOFF) {
        int mb = (lane & 16) ? 8 : 0;
        float bv = offset_b[ch];
        float* op = offs + ((size_t)(b * KOFF + ch) * HH + y) * WW + x0 + mb;
        store_tile(op,      acc0, bv);      // pixels  0..15
        store_tile(op + 16, acc1, bv);      // pixels 16..31
    }
}

// ---------------------------------------------------------------------------
// Main deformable GEMM: M=32 pixels per workgroup, 4 waves.
// Each wave: 2 N-tiles x 2 M-tiles (B fragments register-reused across M).
// K=1152 in two 576-wide chunks (A chunk 32x576 f16 = 36 KB LDS).
// ---------------------------------------------------------------------------
__global__ __launch_bounds__(128) void fam_main(
    const float* __restrict__ u,
    const float* __restrict__ bias,
    const float* __restrict__ offs,
    const _Float16* __restrict__ frag2,
    float* __restrict__ out) {
    __shared__ __align__(16) _Float16 Ash[32 * 576];   // 36 KB
    __shared__ int   midx[32 * 9 * 4];
    __shared__ float mwt[32 * 9 * 4];

    int tile = blockIdx.x;                  // 1024 tiles
    int x0 = (tile & 1) * 32;
    int y  = (tile >> 1) & (HH - 1);
    int b  = tile >> 7;
    int tid = threadIdx.x;

    // Stage 1: bilinear metadata per (pixel p, tap j): 4 corners.
    for (int e = tid; e < 32 * 9; e += 128) {
        int p = e / 9;
        int j = e - p * 9;
        int xg = x0 + p;
        float dy = offs[((size_t)(b * KOFF + 2 * j    ) * HH + y) * WW + xg];
        float dx = offs[((size_t)(b * KOFF + 2 * j + 1) * HH + y) * WW + xg];
        int ky = j / 3, kx = j - ky * 3;
        float py = (float)(y - 1 + ky) + dy;
        float px = (float)(xg - 1 + kx) + dx;
        float y0f = floorf(py), x0f = floorf(px);
        int y0i = (int)y0f, x0i = (int)x0f;
        float wy1 = py - y0f, wy0 = 1.0f - wy1;
        float wx1 = px - x0f, wx0 = 1.0f - wx1;
        int   yis[2] = { y0i, y0i + 1 };
        int   xis[2] = { x0i, x0i + 1 };
        float wys[2] = { wy0, wy1 };
        float wxs[2] = { wx0, wx1 };
#pragma unroll
        for (int q = 0; q < 4; ++q) {
            int yi = yis[q >> 1], xi = xis[q & 1];
            float wq = wys[q >> 1] * wxs[q & 1];
            bool valid = (yi >= 0) && (yi < HH) && (xi >= 0) && (xi < WW);
            midx[e * 4 + q] = iclamp(yi, 0, HH - 1) * WW + iclamp(xi, 0, WW - 1);
            mwt[e * 4 + q]  = valid ? wq : 0.0f;
        }
    }
    __syncthreads();

    int lane = tid & 31;
    int wv   = tid >> 5;
    int m    = lane & 15;
    int aoff = (lane & 16) ? 8 : 0;

    int p     = tid >> 2;                   // pixel 0..31 (A-build role)
    int lane4 = tid & 3;
    const float* ub = u + (size_t)b * CU * PLANE;

    v8f acc00 = {}, acc01 = {}, acc10 = {}, acc11 = {};
    const _Float16* bb0 = frag2 + (size_t)(wv * 2    ) * KB2 * 512;
    const _Float16* bb1 = frag2 + (size_t)(wv * 2 + 1) * KB2 * 512;

    for (int half = 0; half < 2; ++half) {  // 64 u-channels per chunk
        __syncthreads();
        // Stage 2: division-free gather A[p][cl*9+j], c = half*64 + cl.
        for (int j = 0; j < 9; ++j) {
            int mbm = (p * 9 + j) * 4;
            int   i0 = midx[mbm + 0], i1 = midx[mbm + 1];
            int   i2 = midx[mbm + 2], i3 = midx[mbm + 3];
            float w0 = mwt[mbm + 0], w1 = mwt[mbm + 1];
            float w2 = mwt[mbm + 2], w3 = mwt[mbm + 3];
            const float* up = ub + (size_t)(half * 64 + lane4) * PLANE;
            _Float16* adst = Ash + p * 576 + lane4 * 9 + j;
            for (int cc = 0; cc < 16; ++cc) {
                float v = w0 * up[i0] + w1 * up[i1] + w2 * up[i2] + w3 * up[i3];
                *adst = (_Float16)v;
                up   += 4 * PLANE;
                adst += 4 * 9;
            }
        }
        __syncthreads();

        // Stage 3: WMMA, 2x2 micro-tile per wave.
        for (int kb2 = 0; kb2 < 18; ++kb2) {
            int kbg = half * 18 + kb2;
            int ab  = m * 576 + kb2 * 32 + aoff;
            v16h a0 = make_a(Ash + ab);
            v16h a1 = make_a(Ash + ab + 16 * 576);
            v16h bf0 = *(const v16h*)(bb0 + ((size_t)kbg * 32 + lane) * 16);
            v16h bf1 = *(const v16h*)(bb1 + ((size_t)kbg * 32 + lane) * 16);
            acc00 = __builtin_amdgcn_wmma_f32_16x16x32_f16(
                false, a0, false, bf0, (short)0, acc00, false, false);
            acc01 = __builtin_amdgcn_wmma_f32_16x16x32_f16(
                false, a0, false, bf1, (short)0, acc01, false, false);
            acc10 = __builtin_amdgcn_wmma_f32_16x16x32_f16(
                false, a1, false, bf0, (short)0, acc10, false, false);
            acc11 = __builtin_amdgcn_wmma_f32_16x16x32_f16(
                false, a1, false, bf1, (short)0, acc11, false, false);
        }
    }

    // Stage 4: bias + store. D layout: vgpr r, lane<16 -> M=r; lane>=16 -> M=8+r.
    int n  = lane & 15;
    int mb = (lane & 16) ? 8 : 0;
    {
        int oc = wv * 32 + n;                            // tile t0
        float bv = bias[oc];
        float* op = out + ((size_t)(b * CO + oc) * HH + y) * WW + x0 + mb;
        store_tile(op,      acc00, bv);                  // pixels  0..15
        store_tile(op + 16, acc10, bv);                  // pixels 16..31
    }
    {
        int oc = wv * 32 + 16 + n;                       // tile t1
        float bv = bias[oc];
        float* op = out + ((size_t)(b * CO + oc) * HH + y) * WW + x0 + mb;
        store_tile(op,      acc01, bv);
        store_tile(op + 16, acc11, bv);
    }
}

extern "C" void kernel_launch(void* const* d_in, const int* in_sizes, int n_in,
                              void* d_out, int out_size, void* d_ws, size_t ws_size,
                              hipStream_t stream) {
    (void)in_sizes; (void)n_in; (void)out_size; (void)ws_size;
    const float* x        = (const float*)d_in[0];
    const float* u        = (const float*)d_in[1];
    const float* weight   = (const float*)d_in[2];
    const float* bias     = (const float*)d_in[3];
    const float* offset_w = (const float*)d_in[4];
    const float* offset_b = (const float*)d_in[5];
    float* out = (float*)d_out;

    char* ws = (char*)d_ws;
    float*    offs  = (float*)ws;
    _Float16* frag1 = (_Float16*)(ws + (size_t)OFFS_FLOATS * 4);
    _Float16* frag2 = (_Float16*)(ws + (size_t)OFFS_FLOATS * 4 + (size_t)FRAG1_HALFS * 2);

    int prep_blocks = (FRAG1_HALFS + FRAG2_HALFS) / 256;   // 864
    fam_prep_weights<<<prep_blocks, 256, 0, stream>>>(weight, offset_w, frag1, frag2);

    int tiles = BN * HH * (WW / 32);                        // 1024
    fam_offset_conv<<<tiles, 64, 0, stream>>>(x, u, offset_b, frag1, offs);
    fam_main<<<tiles, 128, 0, stream>>>(u, bias, offs, frag2, out);
}
